// state_77962246357349
// MI455X (gfx1250) — compile-verified, matrix operand order checked by reading, not run on hardware
//
#include <hip/hip_runtime.h>
#include <stdint.h>

#define NP      2048
#define NR      128
#define NSTEPS  256
#define MAPW    1024
#define MAPH    1024
#define DBIN    256
#define MAXD    256.0f
#define OCCTH   0.97f
#define BM_WORDS (MAPH * MAPW / 32)   // 32768 words = 128 KB
#define MARCH_LDS (BM_WORDS * 4 + 16 * NR * 4)  // 131072 + 8192 = 139264

typedef float v2f  __attribute__((ext_vector_type(2)));
typedef float v8f  __attribute__((ext_vector_type(8)));
typedef _Float16 v16h __attribute__((ext_vector_type(16)));
typedef unsigned int v4u __attribute__((ext_vector_type(4)));
typedef int v8i __attribute__((ext_vector_type(8)));
typedef int v4i __attribute__((ext_vector_type(4)));

#if __has_builtin(__builtin_amdgcn_tensor_load_to_lds)
#define HAVE_TDM 1
#endif
#if __has_builtin(__builtin_amdgcn_wmma_f32_16x16x4_f32)
#define HAVE_WMMA_F32X4 1
#endif

// ---------------- occupancy bitmap: map_grid > 0.97 -> 1 bit/cell ----------
__global__ void __launch_bounds__(256) k_bitmap(const float* __restrict__ map,
                                                uint32_t* __restrict__ bm) {
  int w = blockIdx.x * blockDim.x + threadIdx.x;
  if (w >= BM_WORDS) return;
  int y  = w >> 5;
  int xb = (w & 31) << 5;
  const float* row = map + y * MAPW + xb;
  uint32_t bits = 0u;
#pragma unroll
  for (int b = 0; b < 32; ++b) bits |= (row[b] > OCCTH) ? (1u << b) : 0u;
  bm[w] = bits;
}

// ---------------- motion update ------------------------------------------
__global__ void __launch_bounds__(256) k_motion(const float* __restrict__ particles,
                                                const float* __restrict__ twist,
                                                const float* __restrict__ mnoise,
                                                float* __restrict__ p) {
  int i = blockIdx.x * blockDim.x + threadIdx.x;
  if (i >= NP) return;
  const float sm[3] = {0.5f, 0.5f, 0.02f};
#pragma unroll
  for (int c = 0; c < 3; ++c)
    p[i * 3 + c] = particles[i * 3 + c] + twist[c] + mnoise[i * 3 + c] * sm[c];
}

// ---------------- ray march + WMMA row-sum -------------------------------
// grid: 128 blocks x 256 threads; block handles 16 particles x 128 rays.
// LDS: [0,128K) occupancy bitmap (TDM-staged), [128K,136K) value tile 16x128.
__global__ void __launch_bounds__(256) k_march(const uint32_t* __restrict__ bm,
                                               const float* __restrict__ p,
                                               const float* __restrict__ obs_angles,
                                               const float* __restrict__ obs_dists,
                                               const float* __restrict__ stab,
                                               float* __restrict__ logl) {
  extern __shared__ char smem[];
  uint32_t* bml   = (uint32_t*)smem;
  float*   valbuf = (float*)(smem + BM_WORDS * 4);

#ifdef HAVE_TDM
  if (threadIdx.x < 32) {  // wave 0 only (uniform branch): one TDM op per block
    uint64_t ga = (uint64_t)(uintptr_t)bm;
    uint32_t la = (uint32_t)(uintptr_t)bml;  // low 32 bits = LDS offset
    // D# group0: count=1, lds_addr, global_addr[56:0], type=2
    v4u g0 = { 1u, la, (uint32_t)ga, (uint32_t)(ga >> 32) | (2u << 30) };
    // D# group1: data_size=4B; tensor_dim0=32768; tensor_dim1=1;
    // tile_dim0=32768 (1D tile); stride0=32768
    v8i g1 = { (int)(2u << 16),        // data_size=2 (4B), mask=0
               (int)0x80000000u,       // tensor_dim0 lo16 @bits63:48
               (int)0x00010000u,       // tensor_dim0 hi=0, tensor_dim1 lo16=1
               (int)0x80000000u,       // tensor_dim1 hi=0, tile_dim0=0x8000
               0,                      // tile_dim1=0, tile_dim2=0
               32768, 0, 0 };          // tensor_dim0_stride=32768
    v4i gz = {0, 0, 0, 0};
#if __clang_major__ >= 23
    v8i gz8 = {0, 0, 0, 0, 0, 0, 0, 0};
    __builtin_amdgcn_tensor_load_to_lds(g0, g1, gz, gz, gz8, 0);
#else
    __builtin_amdgcn_tensor_load_to_lds(g0, g1, gz, gz, 0);
#endif
    __builtin_amdgcn_s_wait_tensorcnt(0);
  }
#else
  for (int i = threadIdx.x; i < BM_WORDS; i += 256) bml[i] = bm[i];
#endif
  __syncthreads();

  // each thread: 1 particle, 8 consecutive rays
  int t       = threadIdx.x;
  int p_local = t >> 4;
  int pg      = blockIdx.x * 16 + p_local;
  float x  = p[pg * 3 + 0];
  float y  = p[pg * 3 + 1];
  float th = p[pg * 3 + 2];
  int rbase = (t & 15) * 8;

#pragma unroll 1
  for (int k = 0; k < 8; ++k) {
    int r = rbase + k;
    float ang = th + obs_angles[r];
    float ct = cosf(ang), st = sinf(ang);
    float dist = MAXD;
#pragma unroll 1
    for (int s = 1; s <= NSTEPS; ++s) {
      float tt = (float)s;
      float px = x + tt * ct;
      float py = y + tt * st;
      int ix = (int)px; ix = ix < 0 ? 0 : (ix > MAPW - 1 ? MAPW - 1 : ix);
      int iy = (int)py; iy = iy < 0 ? 0 : (iy > MAPH - 1 ? MAPH - 1 : iy);
      uint32_t wbits = bml[(iy << 5) + (ix >> 5)];
      if ((wbits >> (ix & 31)) & 1u) { dist = tt; break; }   // first hit = min
    }
    int eb = (int)dist; eb = eb < 0 ? 0 : (eb > DBIN - 1 ? DBIN - 1 : eb);
    float od = obs_dists[r];
    int ob = (int)od; ob = ob < 0 ? 0 : (ob > DBIN - 1 ? DBIN - 1 : ob);
    valbuf[p_local * NR + r] = stab[eb * DBIN + ob];
  }
  __syncthreads();

  // row-sum of 16x128 value tile via WMMA (B = ones => D[m][*] = sum_k A[m][k])
  if (threadIdx.x < 32) {
    int lane = threadIdx.x;
    int m    = lane & 15;
    int koff = (lane >> 4) * 2;            // A 16x4 f32 layout (ISA 7.12.2)
    v8f c = {0.f, 0.f, 0.f, 0.f, 0.f, 0.f, 0.f, 0.f};
#ifdef HAVE_WMMA_F32X4
    v2f bones = {1.0f, 1.0f};
#pragma unroll 1
    for (int j = 0; j < 32; ++j) {
      v2f a;
      a.x = valbuf[m * NR + 4 * j + koff];
      a.y = valbuf[m * NR + 4 * j + koff + 1];
      c = __builtin_amdgcn_wmma_f32_16x16x4_f32(false, a, false, bones,
                                                (short)0, c, false, false);
    }
#else
    // fallback: f16 WMMA 16x16x32 (codegen-confirmed builtin)
    v16h bones;
#pragma unroll
    for (int e = 0; e < 16; ++e) bones[e] = (_Float16)1.0f;
    int b1 = (lane >> 4) ? 8 : 0;       // K bases per 16-bit A layout
    int b2 = (lane >> 4) ? 24 : 16;
#pragma unroll 1
    for (int j = 0; j < 4; ++j) {
      v16h a;
#pragma unroll
      for (int e = 0; e < 16; ++e) {
        int kk = (e < 8) ? (b1 + e) : (b2 + (e - 8));
        a[e] = (_Float16)valbuf[m * NR + 32 * j + kk];
      }
      c = __builtin_amdgcn_wmma_f32_16x16x32_f16(false, a, false, bones,
                                                 (short)0, c, false, false);
    }
#endif
    // D layout: VGPR g holds M=g (lanes 0-15) / M=8+g (lanes 16-31); all N equal
    if ((lane & 15) == 0) {
      int mb = (lane >> 4) * 8;
#pragma unroll
      for (int g = 0; g < 8; ++g) logl[blockIdx.x * 16 + mb + g] = c[g];
    }
  }
}

// ---------------- softmaxes, estimates, cumsum, resample ------------------
__device__ __forceinline__ float red_sum(float v, float* red, int t) {
  red[t] = v; __syncthreads();
  for (int s = 512; s > 0; s >>= 1) { if (t < s) red[t] += red[t + s]; __syncthreads(); }
  float r = red[0]; __syncthreads();
  return r;
}
__device__ __forceinline__ float red_max(float v, float* red, int t) {
  red[t] = v; __syncthreads();
  for (int s = 512; s > 0; s >>= 1) { if (t < s) red[t] = fmaxf(red[t], red[t + s]); __syncthreads(); }
  float r = red[0]; __syncthreads();
  return r;
}

__global__ void __launch_bounds__(1024) k_final(const float* __restrict__ logw,
                                                const float* __restrict__ logl,
                                                const float* __restrict__ p,
                                                const float* __restrict__ resample_u,
                                                const float* __restrict__ noise_after,
                                                float* __restrict__ out) {
  __shared__ float sh_cum[NP];
  __shared__ float sh_red[1024];
  __shared__ float sh_scan[1024];
  int t = threadIdx.x;
  int i0 = 2 * t, i1 = 2 * t + 1;

  float ll0 = logl[i0], ll1 = logl[i1];
  float l0 = logw[i0] + ll0, l1 = logw[i1] + ll1;

  float M = red_max(fmaxf(l0, l1), sh_red, t);
  float e10_0 = expf(10.f * (l0 - M)), e10_1 = expf(10.f * (l1 - M));
  float e0 = expf(l0 - M), e1 = expf(l1 - M);
  float S10 = red_sum(e10_0 + e10_1, sh_red, t);
  float S1  = red_sum(e0 + e1, sh_red, t);
  float we0 = e10_0 / S10, we1 = e10_1 / S10;   // w_est = softmax(10*logits)
  float w0  = e0 / S1,     w1  = e1 / S1;       // w     = softmax(logits)

  float x0 = p[i0 * 3], y0 = p[i0 * 3 + 1], th0 = p[i0 * 3 + 2];
  float x1 = p[i1 * 3], y1 = p[i1 * 3 + 1], th1 = p[i1 * 3 + 2];

  float ex   = red_sum(we0 * x0 + we1 * x1, sh_red, t);
  float ey   = red_sum(we0 * y0 + we1 * y1, sh_red, t);
  float ssin = red_sum(we0 * sinf(th0) + we1 * sinf(th1), sh_red, t);
  float scos = red_sum(we0 * cosf(th0) + we1 * cosf(th1), sh_red, t);
  float conf = red_sum(w0 * ll0 + w1 * ll1, sh_red, t);

  // inclusive scan of per-thread pair sums (Hillis-Steele), then expand
  float ps = w0 + w1;
  sh_scan[t] = ps; __syncthreads();
  for (int off = 1; off < 1024; off <<= 1) {
    float add = (t >= off) ? sh_scan[t - off] : 0.f;
    __syncthreads();
    sh_scan[t] += add;
    __syncthreads();
  }
  float excl = sh_scan[t] - ps;
  sh_cum[i0] = excl + w0;
  sh_cum[i1] = excl + w0 + w1;
  __syncthreads();

  // systematic resample: searchsorted(cum, (i+u)/P), side='left'
  float u = resample_u[0];
  const float sn[3] = {0.2f, 0.2f, 0.01f};
#pragma unroll
  for (int q = 0; q < 2; ++q) {
    int i = 2 * t + q;
    float pos = ((float)i + u) / (float)NP;
    int lo = 0, hi = NP;
    while (lo < hi) { int mid = (lo + hi) >> 1; if (sh_cum[mid] < pos) lo = mid + 1; else hi = mid; }
    int idx = lo > NP - 1 ? NP - 1 : lo;
#pragma unroll
    for (int c = 0; c < 3; ++c)
      out[3 + i * 3 + c] = p[idx * 3 + c] + noise_after[i * 3 + c] * sn[c];
  }
  if (t == 0) {
    out[0] = ex;
    out[1] = ey;
    out[2] = atan2f(ssin, scos);
    out[3 + NP * 3] = conf;
  }
}

extern "C" void kernel_launch(void* const* d_in, const int* in_sizes, int n_in,
                              void* d_out, int out_size, void* d_ws, size_t ws_size,
                              hipStream_t stream) {
  const float* particles    = (const float*)d_in[0];
  const float* log_weights  = (const float*)d_in[1];
  const float* obs_angles   = (const float*)d_in[2];
  const float* obs_dists    = (const float*)d_in[3];
  const float* map_grid     = (const float*)d_in[4];
  const float* sensor_table = (const float*)d_in[5];
  const float* twist        = (const float*)d_in[6];
  const float* motion_noise = (const float*)d_in[7];
  const float* resample_u   = (const float*)d_in[8];
  const float* noise_after  = (const float*)d_in[9];
  float* out = (float*)d_out;

  char* ws = (char*)d_ws;
  uint32_t* bm = (uint32_t*)ws;                                  // 128 KB
  float* p     = (float*)(ws + BM_WORDS * 4);                    // 24 KB
  float* logl  = (float*)(ws + BM_WORDS * 4 + NP * 3 * 4);       // 8 KB

  (void)hipFuncSetAttribute((const void*)k_march,
                            hipFuncAttributeMaxDynamicSharedMemorySize,
                            MARCH_LDS);

  k_bitmap<<<(BM_WORDS + 255) / 256, 256, 0, stream>>>(map_grid, bm);
  k_motion<<<(NP + 255) / 256, 256, 0, stream>>>(particles, twist, motion_noise, p);
  k_march<<<NP / 16, 256, MARCH_LDS, stream>>>(bm, p, obs_angles, obs_dists,
                                               sensor_table, logl);
  k_final<<<1, 1024, 0, stream>>>(log_weights, logl, p, resample_u, noise_after, out);
}
